// MAELDRegLoss_24215025615484
// MI455X (gfx1250) — matrix operand order repaired and checked by reading, not compiled, not run on hardware
//
#include <hip/hip_runtime.h>
#include <hip/hip_bf16.h>

typedef __attribute__((ext_vector_type(16))) __bf16 v16bf;
typedef __attribute__((ext_vector_type(8)))  __bf16 v8bf;
typedef __attribute__((ext_vector_type(8)))  float  v8f;

#define N_PTS   8192
#define DIM     768
#define BROWS   64            // rows per workgroup (4 row-tiles of 16)
#define NWAVES  4
#define NTHREADS (NWAVES * 32)
#define CTILES  (N_PTS / 16)  // 512 column tiles
#define LISTCAP 33            // supports k+1 up to 33; odd stride for bank spread
#define SPAD    17            // padded scratch row stride (floats)

// ---------------------------------------------------------------------------
// Phase 0: split fp32 features into bf16 hi/lo and compute exact fp32 row norms
// ---------------------------------------------------------------------------
__global__ void lid_prep_kernel(const float* __restrict__ X,
                                __bf16* __restrict__ Xhi,
                                __bf16* __restrict__ Xlo,
                                float* __restrict__ norms) {
    __shared__ float red[256];
    const int row = blockIdx.x;
    const float* xr = X + (size_t)row * DIM;
    float s = 0.f;
    for (int i = threadIdx.x; i < DIM; i += 256) {
        float x = xr[i];
        __bf16 h = (__bf16)x;
        float lo = x - (float)h;
        Xhi[(size_t)row * DIM + i] = h;
        Xlo[(size_t)row * DIM + i] = (__bf16)lo;
        s += x * x;
    }
    red[threadIdx.x] = s;
    __syncthreads();
    for (int off = 128; off > 0; off >>= 1) {
        if (threadIdx.x < off) red[threadIdx.x] += red[threadIdx.x + off];
        __syncthreads();
    }
    if (threadIdx.x == 0) norms[row] = red[0];
}

// ---------------------------------------------------------------------------
// Phase 1: fused bf16x2 WMMA distance GEMM + streaming top-(k+1) + LID formula
// ---------------------------------------------------------------------------
__global__ __launch_bounds__(NTHREADS, 1)
void lid_main_kernel(const __bf16* __restrict__ Xhi,
                     const __bf16* __restrict__ Xlo,
                     const float* __restrict__ norms,
                     const int* __restrict__ kptr,
                     float* __restrict__ out) {
    // LDS: A tiles (hi=0, lo=1), row norms, per-wave r2 scratch, candidate lists
    __shared__ __bf16 Asm[2][BROWS * DIM];                 // 192 KB
    __shared__ float  xnorm[BROWS];
    __shared__ float  scratch[NWAVES][BROWS * SPAD];       // ~17 KB
    __shared__ float  cand[NWAVES * BROWS * LISTCAP];      // ~33 KB

    const int tid  = threadIdx.x;
    const int wave = tid >> 5;
    const int lane = tid & 31;
    const int row0 = blockIdx.x * BROWS;
    const int k    = *kptr;        // 20 in reference
    const int kp1  = k + 1;        // keep k+1 smallest (incl. self)

    // ---- preload A tiles (bf16 hi/lo) into LDS, init candidate lists ----
    {
        const v8bf* sh = (const v8bf*)(Xhi + (size_t)row0 * DIM);
        const v8bf* sl = (const v8bf*)(Xlo + (size_t)row0 * DIM);
        v8bf* dh = (v8bf*)&Asm[0][0];
        v8bf* dl = (v8bf*)&Asm[1][0];
        const int nvec = BROWS * DIM / 8;
        for (int i = tid; i < nvec; i += NTHREADS) {
            dh[i] = sh[i];
            dl[i] = sl[i];
        }
        if (tid < BROWS) xnorm[tid] = norms[row0 + tid];
        for (int i = tid; i < NWAVES * BROWS * LISTCAP; i += NTHREADS)
            cand[i] = __builtin_inff();
    }
    __syncthreads();

    // per-owned-row top-(k+1) running state (lane owns rows lane and lane+32)
    float cmax0 = __builtin_inff(), cmax1 = __builtin_inff();
    int   slot0 = 0,                slot1 = 0;
    float* mylist0 = &cand[(wave * BROWS + lane)      * LISTCAP];
    float* mylist1 = &cand[(wave * BROWS + lane + 32) * LISTCAP];

    const int colL = lane & 15;   // N index (B) == M index (A) for this lane
    const int hf   = lane >> 4;   // half-wave selector

    for (int ct = wave; ct < CTILES; ct += NWAVES) {
        const int col0 = ct * 16;
        v8f acc[4] = {};

        // K loop over 3 bf16x2 segments: hi*hi + lo*hi + hi*lo
        for (int seg = 0; seg < 3; ++seg) {
            const __bf16* Bg   = (seg == 2) ? Xlo : Xhi;
            const int     asel = (seg == 1) ? 1 : 0;
            // B operand: lane supplies N=colL, K = k0 + 16*hf + (0..15): 32 contiguous bytes
            const __bf16* Brow = Bg + (size_t)(col0 + colL) * DIM + hf * 16;
#pragma unroll 4
            for (int k0 = 0; k0 < DIM; k0 += 32) {
                v16bf b = *(const v16bf*)(Brow + k0);
#pragma unroll
                for (int t = 0; t < 4; ++t) {
                    // A operand: lane supplies M=colL of row tile t;
                    // K chunks: k0 + 16*hf + {0..7} and k0 + 16*hf + 8 + {0..7}
                    const __bf16* ar =
                        &Asm[asel][(size_t)(t * 16 + colL) * DIM + k0 + hf * 16];
                    union { v16bf v; v8bf h[2]; } a;
                    a.h[0] = *(const v8bf*)(ar);
                    a.h[1] = *(const v8bf*)(ar + 8);
                    acc[t] = __builtin_amdgcn_wmma_f32_16x16x32_bf16(
                        false, a.v, false, b, (short)0, acc[t], false, false);
                }
            }
        }

        // ---- r2 = |x|^2 + |y|^2 - 2 x.y -> per-wave LDS scratch ----
        const float yn = norms[col0 + colL];
#pragma unroll
        for (int t = 0; t < 4; ++t) {
#pragma unroll
            for (int v = 0; v < 8; ++v) {
                const int r = t * 16 + hf * 8 + v;    // D layout: M = v + 8*hf
                float r2 = xnorm[r] + yn - 2.0f * acc[t][v];
                scratch[wave][r * SPAD + colL] = r2;
            }
        }
        __builtin_amdgcn_wave_barrier();   // cross-lane LDS handoff within wave

        // ---- streaming top-(k+1): lane scans 16 candidates for its 2 rows ----
        for (int c = 0; c < 16; ++c) {
            float v0 = scratch[wave][lane * SPAD + c];
            if (v0 < cmax0) {
                mylist0[slot0] = v0;
                float m = -__builtin_inff(); int ms = 0;
                for (int j = 0; j < kp1; ++j) {
                    float x = mylist0[j];
                    if (x > m) { m = x; ms = j; }
                }
                cmax0 = m; slot0 = ms;
            }
            float v1 = scratch[wave][(lane + 32) * SPAD + c];
            if (v1 < cmax1) {
                mylist1[slot1] = v1;
                float m = -__builtin_inff(); int ms = 0;
                for (int j = 0; j < kp1; ++j) {
                    float x = mylist1[j];
                    if (x > m) { m = x; ms = j; }
                }
                cmax1 = m; slot1 = ms;
            }
        }
        __builtin_amdgcn_wave_barrier();
    }

    __syncthreads();

    // ---- merge the NWAVES candidate lists per row; extract k+1 in order ----
    if (tid < BROWS) {
        const int row = tid;
        float sum = 0.f, a_k = 0.f;
        for (int i = 0; i < kp1; ++i) {
            float best = __builtin_inff();
            int bw = 0, bj = 0;
            for (int w = 0; w < NWAVES; ++w) {
                const float* l = &cand[(w * BROWS + row) * LISTCAP];
                for (int j = 0; j < kp1; ++j) {
                    float x = l[j];
                    if (x < best) { best = x; bw = w; bj = j; }
                }
            }
            cand[(bw * BROWS + row) * LISTCAP + bj] = __builtin_inff();
            float d = sqrtf(fmaxf(best, 1e-12f));
            if (i == 0) {
                // a[0]: self distance, excluded
            } else if (i < k) {
                sum += d;              // a[1:k] -> k-1 values
            } else {
                a_k = d;               // a[k]
            }
        }
        float m   = sum / (float)(k - 1);
        float lid = m / (a_k - m);
        out[row0 + row] = -fabsf(logf(lid));
    }
}

// ---------------------------------------------------------------------------
extern "C" void kernel_launch(void* const* d_in, const int* in_sizes, int n_in,
                              void* d_out, int out_size, void* d_ws, size_t ws_size,
                              hipStream_t stream) {
    const float* X    = (const float*)d_in[0];
    const int*   kptr = (const int*)d_in[1];
    float*       out  = (float*)d_out;

    // workspace layout: Xhi (bf16), Xlo (bf16), norms (f32)  -> ~25.2 MB
    char* ws = (char*)d_ws;
    __bf16* Xhi  = (__bf16*)ws;
    __bf16* Xlo  = (__bf16*)(ws + (size_t)N_PTS * DIM * 2);
    float*  norms = (float*)(ws + (size_t)N_PTS * DIM * 4);

    lid_prep_kernel<<<N_PTS, 256, 0, stream>>>(X, Xhi, Xlo, norms);
    lid_main_kernel<<<N_PTS / BROWS, NTHREADS, 0, stream>>>(Xhi, Xlo, norms, kptr, out);
}
